// RegressionTransformer2_74045236183100
// MI455X (gfx1250) — matrix operand order repaired and checked by reading, not compiled, not run on hardware
//
#include <hip/hip_runtime.h>
#include <hip/hip_fp16.h>
#include <hip/hip_bf16.h>

typedef _Float16 v16h __attribute__((ext_vector_type(16)));
typedef _Float16 v8h  __attribute__((ext_vector_type(8)));
typedef float    v8f  __attribute__((ext_vector_type(8)));

#define B_    256
#define N_    512
#define G_    117
#define CH_   25
#define K_    625
#define NKC   20          // ceil(640/32) K-chunks of 32
#define NTILES 8          // 128 padded genes / 16
#define ROWS  (B_*N_)     // 131072 rows of the big GEMM
#define MTILES (ROWS/16)  // 8192 blocks

#define CELL_BASE ((size_t)(B_*G_))
#define OUT1_BASE ((size_t)(B_*G_) + (size_t)B_*N_*G_)

// ---------------------------------------------------------------- zero spot
__global__ void rt2_zero_spot(float* __restrict__ out) {
    int i = blockIdx.x * blockDim.x + threadIdx.x;
    if (i < B_*G_) out[i] = 0.0f;
}

// ------------------------------------------- patch conv + global embedding
// spot_images (B,1,28,28) -> conv k5 s5 -> (3,5,5) -> tokens (25,3) -> flat 75
// glob[b] = ELU(flat . W_glob + b_glob)
__global__ void rt2_glob(const float* __restrict__ spot_images,
                         const float* __restrict__ W_patch,
                         const float* __restrict__ b_patch,
                         const float* __restrict__ W_glob,
                         const float* __restrict__ b_glob,
                         float* __restrict__ glob) {
    int b = blockIdx.x * blockDim.x + threadIdx.x;
    if (b >= B_) return;
    const float* img = spot_images + (size_t)b * 784;
    float acc = b_glob[0];
    for (int i = 0; i < 5; ++i)
        for (int j = 0; j < 5; ++j) {
            int t = i * 5 + j;
            for (int c = 0; c < 3; ++c) {
                float s = b_patch[c];
                for (int ky = 0; ky < 5; ++ky)
                    for (int kx = 0; kx < 5; ++kx)
                        s += img[(i*5+ky)*28 + (j*5+kx)] * W_patch[c*25 + ky*5 + kx];
                acc += s * W_glob[t*3 + c];   // token-major, channel-minor (transpose)
            }
        }
    glob[b] = acc > 0.0f ? acc : (expf(acc) - 1.0f);   // ELU(alpha=1)
}

// --------------------------- repack W_reg (117x625 f32) into f16 B-operand
// Layout per (n_tile, k_chunk): 32 lanes x 16 halves, contiguous per lane.
// Lane l holds column n = l&15; half h holds K = kc*32 + (l>=16 ? 8:0) + (h&7) + (h>=8 ? 16:0).
__global__ void rt2_pack_b(const float* __restrict__ W_reg,
                           _Float16* __restrict__ Bpack) {
    int idx = blockIdx.x * blockDim.x + threadIdx.x;
    if (idx >= NTILES * NKC * 32 * 16) return;
    int h    = idx & 15;
    int lane = (idx >> 4) & 31;
    int kc   = (idx >> 9) % NKC;
    int nt   = idx / (NKC * 32 * 16);
    int g = nt * 16 + (lane & 15);
    int k = kc * 32 + ((lane >= 16) ? 8 : 0) + (h & 7) + ((h >> 3) << 4);
    float v = (g < G_ && k < K_) ? W_reg[(size_t)g * K_ + k] : 0.0f;
    Bpack[idx] = (_Float16)v;
}

// -------------------------------- fused conv -> WMMA GEMM -> epilogue
// One block = one 16-row M-tile (all rows share batch b). 8 waves, each wave
// owns one 16-wide N-tile of the 128-padded gene dim.
__global__ __launch_bounds__(256) void rt2_fused(
        const float* __restrict__ images,
        const float* __restrict__ W_emph,
        const float* __restrict__ b_emph,
        const float* __restrict__ b_reg,
        const _Float16* __restrict__ Bpack,
        const float* __restrict__ glob,
        float* __restrict__ out) {
    // A tile in WMMA-friendly row-major f16; stride 648 halves (1296 B = 81*16B)
    // rotates LDS banks by 4 dwords/row -> conflict-light b128 lane reads.
    __shared__ __align__(16) _Float16 sA[16 * 648];
    __shared__ float sImg[16 * 50];
    __shared__ float sW[CH_ * 9];
    __shared__ float sBi[CH_];

    const int tid    = threadIdx.x;
    const int tile   = blockIdx.x;
    const int m_base = tile * 16;
    const int b      = tile >> 5;          // 512/16 = 32 M-tiles per batch

    if (tid < CH_ * 9) sW[tid]  = W_emph[tid];
    if (tid < CH_)     sBi[tid] = b_emph[tid];
    for (int i = tid; i < 16 * 49; i += 256) {
        int r = i / 49, p = i % 49;
        sImg[r * 50 + p] = images[(size_t)(m_base + r) * 49 + p];
    }
    __syncthreads();

    // 3x3 valid conv (1->25ch) + ReLU, packed f16 into sA[row][k], k = ch*25+pos
    for (int i = tid; i < 16 * 625; i += 256) {
        int r = i / 625, k = i % 625;
        int ch = k / 25, pos = k % 25;
        int oy = pos / 5, ox = pos % 5;
        const float* ip = &sImg[r * 50 + oy * 7 + ox];
        const float* wp = &sW[ch * 9];
        float a = sBi[ch];
        a += ip[0]  * wp[0] + ip[1]  * wp[1] + ip[2]  * wp[2];
        a += ip[7]  * wp[3] + ip[8]  * wp[4] + ip[9]  * wp[5];
        a += ip[14] * wp[6] + ip[15] * wp[7] + ip[16] * wp[8];
        a = a > 0.0f ? a : 0.0f;
        sA[r * 648 + k] = (_Float16)a;
    }
    for (int i = tid; i < 16 * 15; i += 256) {         // zero K pad 625..639
        int r = i / 15, k = 625 + i % 15;
        sA[r * 648 + k] = (_Float16)0.0f;
    }
    __syncthreads();

    const int w    = tid >> 5;        // wave -> N-tile
    const int lane = tid & 31;
    const int mrow = lane & 15;
    const int alo  = (lane >= 16) ? 8 : 0;   // A-layout: K half-offset per lane group

    v8f acc = {0.f, 0.f, 0.f, 0.f, 0.f, 0.f, 0.f, 0.f};
    const _Float16* bptr = Bpack + ((size_t)(w * NKC) * 32 + lane) * 16;

    #pragma unroll 4
    for (int kc = 0; kc < NKC; ++kc) {
        union { v16h v; v8h h[2]; } ua, ub;
        const _Float16* ap = &sA[mrow * 648 + kc * 32 + alo];
        ua.h[0] = *(const v8h*)(ap);          // K = base .. base+7   (ds_load_b128)
        ua.h[1] = *(const v8h*)(ap + 16);     // K = base+16 .. base+23
        ub.h[0] = *(const v8h*)(bptr);        // pre-swizzled B operand
        ub.h[1] = *(const v8h*)(bptr + 8);
        bptr += 32 * 16;                      // next K-chunk
        acc = __builtin_amdgcn_wmma_f32_16x16x32_f16(
                  false, ua.v, false, ub.v, (short)0, acc, false, false);
    }

    // Epilogue: D[m][n] lives in acc[r] of lane (n + 16*(m>=8)), m = r (+8)
    const float gl = glob[b];
    const int   g  = w * 16 + mrow;
    const bool  gv = (g < G_);
    const float br = gv ? b_reg[g] : 0.0f;
    float colsum = 0.0f;
    #pragma unroll
    for (int r = 0; r < 8; ++r) {
        const int m = m_base + alo + r;
        float cv = acc[r] + br + gl;
        cv = cv > 0.0f ? cv : 0.0f;           // cell = relu(out + glob)
        colsum += cv;
        if (gv) {
            out[CELL_BASE + (size_t)m * G_ + g] = cv;
            out[OUT1_BASE + (size_t)m * G_ + g] = gl;   // broadcast glob
        }
    }
    // combine the two lane-halves of each column, then one atomic per (b,g)
    colsum += __shfl_xor(colsum, 16, 32);
    if (gv && lane < 16)
        unsafeAtomicAdd(&out[(size_t)b * G_ + g], colsum);   // spot = sum over cells
}

extern "C" void kernel_launch(void* const* d_in, const int* in_sizes, int n_in,
                              void* d_out, int out_size, void* d_ws, size_t ws_size,
                              hipStream_t stream) {
    (void)in_sizes; (void)n_in; (void)out_size; (void)ws_size;
    const float* images      = (const float*)d_in[0];
    const float* spot_images = (const float*)d_in[1];
    /* d_in[2] gene_expression: unused by the reference graph */
    const float* W_emph  = (const float*)d_in[3];
    const float* b_emph  = (const float*)d_in[4];
    const float* W_reg   = (const float*)d_in[5];
    const float* b_reg   = (const float*)d_in[6];
    const float* W_patch = (const float*)d_in[7];
    const float* b_patch = (const float*)d_in[8];
    const float* W_glob  = (const float*)d_in[9];
    const float* b_glob  = (const float*)d_in[10];
    float* out = (float*)d_out;

    float*    glob  = (float*)d_ws;                       // 256 f32
    _Float16* Bpack = (_Float16*)((char*)d_ws + 1024);    // 81920 f16 (160 KB)

    rt2_zero_spot<<<(B_*G_ + 255) / 256, 256, 0, stream>>>(out);
    rt2_glob<<<1, 256, 0, stream>>>(spot_images, W_patch, b_patch, W_glob, b_glob, glob);
    rt2_pack_b<<<(NTILES*NKC*32*16 + 255) / 256, 256, 0, stream>>>(W_reg, Bpack);
    rt2_fused<<<MTILES, 256, 0, stream>>>(images, W_emph, b_emph, b_reg, Bpack, glob, out);
}
